// FixedPointAttention_56942676410839
// MI455X (gfx1250) — matrix operand order repaired and checked by reading, not compiled, not run on hardware
//
#include <hip/hip_runtime.h>

// FixedPointAttention for MI455X (gfx1250, wave32, WMMA f16 16x16x32).
//
// Round 3: stage dense-GEMM tiles with the Tensor Data Mover
// (TENSOR_LOAD_TO_LDS, TENSORcnt, double-buffered; D# pad fields reproduce the
// padded LDS row stride), keep GLOBAL_LOAD_ASYNC_TO_LDS_B128 staging in the
// attention kernel, WMMA f16 compute unchanged.

typedef __attribute__((ext_vector_type(16))) _Float16 v16h;
typedef __attribute__((ext_vector_type(8)))  float    v8f;
typedef __attribute__((ext_vector_type(4)))  unsigned int v4u;
typedef __attribute__((ext_vector_type(8)))  int      v8i;
typedef __attribute__((ext_vector_type(4)))  int      v4i;

#define EMBED 512
#define SEQ   1024
#define BATCH 8
#define HEADS 16
#define HDIM  32
#define ROWS  (BATCH * SEQ)   /* 8192 */
#define QKV_N (3 * EMBED)     /* 1536 */

#ifndef __has_builtin
#define __has_builtin(x) 0
#endif
#if __has_builtin(__builtin_amdgcn_tensor_load_to_lds)
#define HAVE_TDM 1
#else
#define HAVE_TDM 0
#endif

// fixed-point quantize: round(v*256) clipped to int16, /256 (RNE)
__device__ __forceinline__ float fpq(float v) {
  float r = rintf(v * 256.0f);
  r = fmaxf(-32768.0f, fminf(32767.0f, r));
  return r * 0.00390625f;
}

// ---- async memory -> LDS (ASYNCcnt) ---------------------------------------
// For LDS-aperture flat addresses the LDS address is the low 32 address bits, so the low 32 bits
// of a generic __shared__ pointer are the LDS byte address the VDST expects.
__device__ __forceinline__ void async_ld16(void* lds_dst, const void* gsrc) {
  asm volatile("global_load_async_to_lds_b128 %0, %1, off"
               :
               : "v"((unsigned)(uintptr_t)lds_dst), "v"(gsrc)
               : "memory");
}
__device__ __forceinline__ void wait_async_le3() {
  asm volatile("s_wait_asynccnt 0x3" ::: "memory");
}
__device__ __forceinline__ void wait_async_0() {
  asm volatile("s_wait_asynccnt 0x0" ::: "memory");
}
__device__ __forceinline__ void wait_tensor_le2() {
  asm volatile("s_wait_tensorcnt 0x2" ::: "memory");
}
__device__ __forceinline__ void wait_tensor_0() {
  asm volatile("s_wait_tensorcnt 0x0" ::: "memory");
}

#if HAVE_TDM
// ---- Tensor Data Mover: 2-D f16 tile, global -> LDS -----------------------
// Loads tile_h rows x 32 halves (64B = 16 dwords) per row; pad_enable inserts
// 4 pad dwords after every 16 stored dwords => LDS row stride 40 halves,
// matching the WMMA fragment loaders' layout.
// D# per CDNA5 ISA ch.8: group0 {count, lds_addr, global_addr, type=2},
// group1 {data_size=2B, pad ctl, tensor dims, tile dims, dim0 stride}.
__device__ __forceinline__ void tdm_load_tile32(
    void* lds_dst, const void* gsrc, unsigned tile_h,
    unsigned tensor_w /*row length in halves*/) {
  unsigned lds = (unsigned)(uintptr_t)lds_dst;
  unsigned long long ga = (unsigned long long)(uintptr_t)gsrc;
  v4u g0;
  g0[0] = 1u;                                              // count=1, user mode
  g0[1] = lds;                                             // lds_addr
  g0[2] = (unsigned)ga;                                    // global_addr[31:0]
  g0[3] = (unsigned)((ga >> 32) & 0x01FFFFFFu) | (2u << 30);  // ga[56:32]|type=2
  v8i g1;
  g1[0] = (int)((1u << 16)        // data_size = 1 -> 2 bytes
              | (1u << 20)        // pad_enable
              | (3u << 22)        // pad_interval: 16 dwords
              | (3u << 25));      // pad_amount:   4 dwords
  g1[1] = (int)((tensor_w & 0xFFFFu) << 16);               // tensor_dim0 lo16
  g1[2] = (int)((tensor_w >> 16) | (0xFFFFu << 16));       // dim0 hi | dim1 lo
  g1[3] = (int)((0x000Fu) | (32u << 16));                  // dim1 hi | tile_dim0=32
  g1[4] = (int)(tile_h & 0xFFFFu);                         // tile_dim1 (dim2=0)
  g1[5] = (int)tensor_w;                                   // dim0_stride[31:0]
  g1[6] = 0;                                               // stride hi, dim1_stride lo
  g1[7] = 0;
  v4i z4 = {0, 0, 0, 0};
#if defined(__clang_major__) && __clang_major__ >= 23
  v8i z8 = {0, 0, 0, 0, 0, 0, 0, 0};
  __builtin_amdgcn_tensor_load_to_lds(g0, g1, z4, z4, z8, 0);
#else
  __builtin_amdgcn_tensor_load_to_lds(g0, g1, z4, z4, 0);
#endif
}
#endif  // HAVE_TDM

// ---- WMMA fragment loaders (wave32, 16-bit operands) -----------------------
// A fragment: 16(M) x 32(K) f16 from row-major tile, leading dim `ld` halves.
// lanes 0-15: rows, K {0..7, 16..23}; lanes 16-31: rows, K {8..15, 24..31}.
__device__ __forceinline__ v16h load_frag_a(const _Float16* p, int ld) {
  const int lane = threadIdx.x & 31;
  const int row  = lane & 15;
  const int kb   = (lane & 16) ? 8 : 0;
  const _Float16* r = p + row * ld + kb;
  v16h f;
#pragma unroll
  for (int i = 0; i < 8; ++i) f[i] = r[i];
#pragma unroll
  for (int i = 0; i < 8; ++i) f[8 + i] = r[16 + i];
  return f;
}

// B fragment: 32(K) x 16(N) f16; source stored transposed as Bt[N][K].
// lanes 0-15: N=lane, K {0..7, 16..23}; lanes 16-31: N=lane-16, K {8..15, 24..31}.
__device__ __forceinline__ v16h load_frag_b(const _Float16* p, int ld) {
  const int lane = threadIdx.x & 31;
  const int col  = lane & 15;
  const int kb   = (lane & 16) ? 8 : 0;
  const _Float16* r = p + col * ld + kb;
  v16h f;
#pragma unroll
  for (int i = 0; i < 8; ++i) f[i] = r[i];
#pragma unroll
  for (int i = 0; i < 8; ++i) f[8 + i] = r[16 + i];
  return f;
}

// ---- elementwise prep kernels ---------------------------------------------
__global__ void f32_to_f16_kernel(const float* __restrict__ s,
                                  _Float16* __restrict__ d, int n) {
  int i = blockIdx.x * blockDim.x + threadIdx.x;
  if (i < n) d[i] = (_Float16)s[i];
}

// src [R][C] f32 -> dst [C][R] f16 (dst[n][k] = src[k][n])
__global__ void transpose_to_f16_kernel(const float* __restrict__ src,
                                        _Float16* __restrict__ dst,
                                        int R, int C) {
  int idx = blockIdx.x * blockDim.x + threadIdx.x;
  if (idx < R * C) {
    int n = idx / R;
    int k = idx - n * R;
    dst[idx] = (_Float16)src[(size_t)k * C + n];
  }
}

// ---- per-lane async tile staging (fallback + attention) -------------------
__device__ __forceinline__ void stage_panel_async(
    const _Float16* __restrict__ Aglob, const _Float16* __restrict__ Bglob,
    _Float16* Abuf, _Float16* Bbuf, int tid) {
#pragma unroll
  for (int t = 0; t < 2; ++t) {
    int idx = tid + t * 256;
    int r = idx >> 2, c = idx & 3;
    async_ld16(Abuf + r * 40 + c * 8, Aglob + (size_t)r * EMBED + c * 8);
  }
  {
    int r = tid >> 2, c = tid & 3;
    async_ld16(Bbuf + r * 40 + c * 8, Bglob + (size_t)r * EMBED + c * 8);
  }
}

// ---- shared GEMM K-loop body ----------------------------------------------
// Stages panel p+1 (TDM if available, else per-lane async), retires panel p,
// then runs 4 WMMAs per wave on the current buffers.
__device__ __forceinline__ void gemm_mainloop(
    const _Float16* Ag, const _Float16* Bg, int brows,
    _Float16 (*As)[128 * 40], _Float16 (*Bs)[64 * 40],
    int tid, int w, v8f acc[4]) {
  const int NP = EMBED / 32;
#if HAVE_TDM
  if (tid < 32) {  // wave 0 drives the Tensor Data Mover (EXEC-independent)
    tdm_load_tile32(As[0], Ag, 128, EMBED);
    tdm_load_tile32(Bs[0], Bg, brows, EMBED);
  }
#else
  stage_panel_async(Ag, Bg, As[0], Bs[0], tid);
#endif
  for (int p = 0; p < NP; ++p) {
    const int buf = p & 1;
#if HAVE_TDM
    if (tid < 32) {
      if (p + 1 < NP) {
        tdm_load_tile32(As[buf ^ 1], Ag + (p + 1) * 32, 128, EMBED);
        tdm_load_tile32(Bs[buf ^ 1], Bg + (p + 1) * 32, brows, EMBED);
        wait_tensor_le2();  // in-order: oldest 2 (panel p) now in LDS
      } else {
        wait_tensor_0();
      }
    }
#else
    if (p + 1 < NP) {
      stage_panel_async(Ag + (p + 1) * 32, Bg + (p + 1) * 32,
                        As[buf ^ 1], Bs[buf ^ 1], tid);
      wait_async_le3();
    } else {
      wait_async_0();
    }
#endif
    __syncthreads();

    v16h a = load_frag_a(As[buf] + w * 16 * 40, 40);
#pragma unroll
    for (int nt = 0; nt < 4; ++nt) {
      v16h b = load_frag_b(Bs[buf] + nt * 16 * 40, 40);
      acc[nt] = __builtin_amdgcn_wmma_f32_16x16x32_f16(
          false, a, false, b, (short)0, acc[nt], false, false);
    }
    __syncthreads();
  }
}

// ---- QKV GEMM: [8192 x 512] x [512 x 1536] + bias, quantize, scatter ------
__global__ __launch_bounds__(256) void qkv_gemm_kernel(
    const _Float16* __restrict__ xh, const _Float16* __restrict__ WT,
    const float* __restrict__ bias,
    _Float16* __restrict__ qh, _Float16* __restrict__ kh,
    _Float16* __restrict__ vh) {
  __shared__ __align__(16) _Float16 As[2][128 * 40];
  __shared__ __align__(16) _Float16 Bs[2][64 * 40];

  const int bid = blockIdx.x;
  const int nb  = bid % (QKV_N / 64);
  const int mb  = bid / (QKV_N / 64);
  const int m0  = mb * 128, n0 = nb * 64;
  const int tid = threadIdx.x, lane = tid & 31, w = tid >> 5;

  v8f acc[4] = {};
  gemm_mainloop(xh + (size_t)m0 * EMBED, WT + (size_t)n0 * EMBED, 64,
                As, Bs, tid, w, acc);

  // epilogue: bias + quantize + scatter into q/k/v [B,H,S,D] f16
#pragma unroll
  for (int nt = 0; nt < 4; ++nt) {
#pragma unroll
    for (int i = 0; i < 8; ++i) {
      int row = m0 + w * 16 + i + ((lane & 16) ? 8 : 0);
      int col = n0 + nt * 16 + (lane & 15);
      float v = fpq(acc[nt][i] + bias[col]);
      int which = col >> 9;        // 0:q 1:k 2:v
      int e = col & (EMBED - 1);
      int hh = e >> 5, d = e & (HDIM - 1);
      int bb = row >> 10, s = row & (SEQ - 1);
      _Float16* dst = (which == 0) ? qh : ((which == 1) ? kh : vh);
      dst[((((size_t)bb * HEADS + hh) * SEQ) + s) * HDIM + d] = (_Float16)v;
    }
  }
}

// ---- Flash attention: one (b,h) x 128-query block per workgroup -----------
__global__ __launch_bounds__(256) void attn_kernel(
    const _Float16* __restrict__ qh, const _Float16* __restrict__ kh,
    const _Float16* __restrict__ vh, _Float16* __restrict__ ctx) {
  __shared__ __align__(16) _Float16 Qs[128 * 40];   // q tile  [128][32]+pad
  __shared__ __align__(16) _Float16 Ks[32 * 40];    // k block [32 keys][32 d]
  __shared__ __align__(16) _Float16 Vt[32 * 40];    // vT      [32 d][32 keys]
  __shared__ float Sc[128 * 33];                    // scores f32
  __shared__ __align__(16) _Float16 Ps[128 * 40];   // quantized probs f16
  __shared__ float rowM[128], rowL[128], rowA[128];

  const int bh = blockIdx.x >> 3;     // b*16+h
  const int qb = blockIdx.x & 7;
  const _Float16* Q = qh + (size_t)bh * SEQ * HDIM + (size_t)qb * 128 * HDIM;
  const _Float16* K = kh + (size_t)bh * SEQ * HDIM;
  const _Float16* V = vh + (size_t)bh * SEQ * HDIM;

  const int tid = threadIdx.x, lane = tid & 31, w = tid >> 5;
  const float scale = 0.17677669529663687f;  // 1/sqrt(32)

  // stage the 128x32 query tile asynchronously (2 x b128 per thread)
#pragma unroll
  for (int t = 0; t < 2; ++t) {
    int idx = tid + t * 256;
    int r = idx >> 2, c = idx & 3;
    async_ld16(Qs + r * 40 + c * 8, Q + (size_t)r * 32 + c * 8);
  }
  if (tid < 128) { rowM[tid] = -3.0e38f; rowL[tid] = 0.0f; }

  v8f o0 = {}, o1 = {};

  for (int kb = 0; kb < SEQ / 32; ++kb) {
    __syncthreads();  // previous iteration's readers of Ks/Vt/Ps are done
    // stage K block [32][32] asynchronously (128 x b128 chunks)
    if (tid < 128) {
      int r = tid >> 2, c = tid & 3;
      async_ld16(Ks + r * 40 + c * 8,
                 K + (size_t)(kb * 32 + r) * 32 + c * 8);
    }
    // stage V block transposed: Vt[d][key] (elementwise)
#pragma unroll
    for (int t = 0; t < 4; ++t) {
      int idx = tid + t * 256;
      int j = idx >> 5, d = idx & 31;
      Vt[d * 40 + j] = V[(size_t)(kb * 32 + j) * 32 + d];
    }
    if (kb + 1 < SEQ / 32)  // prefetch a line of the next V block
      __builtin_prefetch(V + (size_t)((kb + 1) * 32 + (tid & 31)) * 32, 0, 1);
    wait_async_0();         // own async loads (K now; Q on first iteration)
    __syncthreads();

    // scores: wave w -> q rows [16w,16w+16). K=32=D: one WMMA per key tile.
    {
      v16h a = load_frag_a(Qs + w * 16 * 40, 40);
#pragma unroll
      for (int kt = 0; kt < 2; ++kt) {
        v16h b = load_frag_b(Ks + kt * 16 * 40, 40);
        v8f z = {};
        v8f c = __builtin_amdgcn_wmma_f32_16x16x32_f16(
            false, a, false, b, (short)0, z, false, false);
#pragma unroll
        for (int i = 0; i < 8; ++i) {
          int r  = w * 16 + i + ((lane & 16) ? 8 : 0);
          int cc = kt * 16 + (lane & 15);
          Sc[r * 33 + cc] = c[i] * scale;
        }
      }
    }
    __syncthreads();

    // online softmax per row (128 rows, one thread each), quantized probs
    if (tid < 128) {
      int r = tid;
      float m = rowM[r];
      float bm = m;
#pragma unroll 8
      for (int c = 0; c < 32; ++c) bm = fmaxf(bm, Sc[r * 33 + c]);
      float al = __expf(m - bm);
      float l = rowL[r] * al;
#pragma unroll 8
      for (int c = 0; c < 32; ++c) {
        float p = __expf(Sc[r * 33 + c] - bm);
        p = rintf(p * 256.0f) * 0.00390625f;  // fixed-point quantize (p<=1)
        l += p;
        Ps[r * 40 + c] = (_Float16)p;
      }
      rowM[r] = bm; rowL[r] = l; rowA[r] = al;
    }
    __syncthreads();

    // rescale running output and accumulate P @ V
    {
#pragma unroll
      for (int i = 0; i < 8; ++i) {
        int r = w * 16 + i + ((lane & 16) ? 8 : 0);
        float al = rowA[r];
        o0[i] *= al;
        o1[i] *= al;
      }
      v16h ap = load_frag_a(Ps + w * 16 * 40, 40);
      v16h b0 = load_frag_b(Vt, 40);
      v16h b1 = load_frag_b(Vt + 16 * 40, 40);
      o0 = __builtin_amdgcn_wmma_f32_16x16x32_f16(false, ap, false, b0,
                                                  (short)0, o0, false, false);
      o1 = __builtin_amdgcn_wmma_f32_16x16x32_f16(false, ap, false, b1,
                                                  (short)0, o1, false, false);
    }
  }
  __syncthreads();

  // normalize and write context [B,S,E] f16 (e = h*32 + d)
  const int b = bh >> 4, h = bh & 15;
#pragma unroll
  for (int i = 0; i < 8; ++i) {
    int r = w * 16 + i + ((lane & 16) ? 8 : 0);
    float inv = 1.0f / rowL[r];
    int s = qb * 128 + r;
    size_t base = ((size_t)b * SEQ + s) * EMBED + h * HDIM;
    ctx[base + (lane & 15)]      = (_Float16)(o0[i] * inv);
    ctx[base + 16 + (lane & 15)] = (_Float16)(o1[i] * inv);
  }
}

// ---- Output GEMM: [8192 x 512] x [512 x 512] + bias, quantize, f32 out ----
__global__ __launch_bounds__(256) void o_gemm_kernel(
    const _Float16* __restrict__ ch, const _Float16* __restrict__ WT,
    const float* __restrict__ bias, float* __restrict__ out) {
  __shared__ __align__(16) _Float16 As[2][128 * 40];
  __shared__ __align__(16) _Float16 Bs[2][64 * 40];

  const int bid = blockIdx.x;
  const int nb  = bid % (EMBED / 64);
  const int mb  = bid / (EMBED / 64);
  const int m0  = mb * 128, n0 = nb * 64;
  const int tid = threadIdx.x, lane = tid & 31, w = tid >> 5;

  v8f acc[4] = {};
  gemm_mainloop(ch + (size_t)m0 * EMBED, WT + (size_t)n0 * EMBED, 64,
                As, Bs, tid, w, acc);

#pragma unroll
  for (int nt = 0; nt < 4; ++nt) {
#pragma unroll
    for (int i = 0; i < 8; ++i) {
      int row = m0 + w * 16 + i + ((lane & 16) ? 8 : 0);
      int col = n0 + nt * 16 + (lane & 15);
      out[(size_t)row * EMBED + col] = fpq(acc[nt][i] + bias[col]);
    }
  }
}

// ---- host-side launch ------------------------------------------------------
extern "C" void kernel_launch(void* const* d_in, const int* in_sizes, int n_in,
                              void* d_out, int out_size, void* d_ws,
                              size_t ws_size, hipStream_t stream) {
  (void)in_sizes; (void)n_in; (void)out_size; (void)ws_size;
  const float* x    = (const float*)d_in[0];
  const float* Wqkv = (const float*)d_in[1];
  const float* bqkv = (const float*)d_in[2];
  const float* Wo   = (const float*)d_in[3];
  const float* bo   = (const float*)d_in[4];
  float* out = (float*)d_out;

  // workspace layout (f16 elements): ~44 MB total
  _Float16* xh    = (_Float16*)d_ws;
  _Float16* WqkvT = xh + (size_t)ROWS * EMBED;           // 1536 x 512
  _Float16* WoT   = WqkvT + (size_t)QKV_N * EMBED;       // 512 x 512
  _Float16* qh    = WoT + (size_t)EMBED * EMBED;         // [B,H,S,D]
  _Float16* kh    = qh + (size_t)ROWS * EMBED;
  _Float16* vh    = kh + (size_t)ROWS * EMBED;
  _Float16* ctx   = vh + (size_t)ROWS * EMBED;           // [B,S,E]

  f32_to_f16_kernel<<<(ROWS * EMBED) / 256, 256, 0, stream>>>(x, xh,
                                                              ROWS * EMBED);
  transpose_to_f16_kernel<<<(EMBED * QKV_N) / 256, 256, 0, stream>>>(
      Wqkv, WqkvT, EMBED, QKV_N);
  transpose_to_f16_kernel<<<(EMBED * EMBED) / 256, 256, 0, stream>>>(
      Wo, WoT, EMBED, EMBED);

  qkv_gemm_kernel<<<(ROWS / 128) * (QKV_N / 64), 256, 0, stream>>>(
      xh, WqkvT, bqkv, qh, kh, vh);

  attn_kernel<<<BATCH * HEADS * (SEQ / 128), 256, 0, stream>>>(qh, kh, vh,
                                                               ctx);

  o_gemm_kernel<<<(ROWS / 128) * (EMBED / 64), 256, 0, stream>>>(ctx, WoT, bo,
                                                                 out);
}
